// CompressedModel_17239998726556
// MI455X (gfx1250) — matrix operand order repaired and checked by reading, not compiled, not run on hardware
//
#include <hip/hip_runtime.h>
#include <cmath>

typedef __attribute__((ext_vector_type(16))) _Float16 v16h;
typedef __attribute__((ext_vector_type(8)))  _Float16 v8h;
typedef __attribute__((ext_vector_type(8)))  float    v8f;

#define B_      4
#define T_      4096
#define C_      768
#define R_      204            /* floor(T - T*0.95) */
#define TOUT_   (T_ - R_)      /* 3892 */
#define NPROT_  (T_ - 2*R_)    /* 3688 */
#define NSPLIT_ 4              /* N-dimension split for occupancy */
#define NCHUNK_ (T_ / NSPLIT_) /* 1024 columns per workgroup */

// ---------------------------------------------------------------------------
// 1) Row L2-normalize x (fp32) -> xn (fp16) for WMMA
// ---------------------------------------------------------------------------
__global__ __launch_bounds__(256) void pitome_norm_kernel(const float* __restrict__ x,
                                                          _Float16* __restrict__ xn) {
    __shared__ float red[256];
    const int row = blockIdx.x;                 // b*T + t
    const int tid = threadIdx.x;
    const float* xr = x + (size_t)row * C_;
    float s = 0.f;
#pragma unroll
    for (int e = 0; e < 3; ++e) {
        float v = xr[tid + e * 256];
        s += v * v;
    }
    red[tid] = s;
    __syncthreads();
    for (int off = 128; off > 0; off >>= 1) {
        if (tid < off) red[tid] += red[tid + off];
        __syncthreads();
    }
    const float inv = 1.0f / fmaxf(sqrtf(red[0]), 1e-12f);
    _Float16* xo = xn + (size_t)row * C_;
#pragma unroll
    for (int e = 0; e < 3; ++e) {
        int c = tid + e * 256;
        xo[c] = (_Float16)(xr[c] * inv);
    }
}

// ---------------------------------------------------------------------------
// 2) Fused WMMA GEMM: partial[s][b][i] = sum_{j in chunk s} elu((xn_i.xn_j - m)*100)
//    Grid: B * (T/128) * NSPLIT blocks of 256 (8 waves). Each wave owns a
//    16-row M-tile (A fragments register-resident for all of K=768).
//    B column tiles (16x768 f16 = 24KB, memory-contiguous) are staged into
//    LDS once per workgroup, double-buffered, and read by all 8 waves via
//    ds_load_b128 -- removing the 8x-redundant vector-memory traffic.
// ---------------------------------------------------------------------------
__global__ __launch_bounds__(256) void pitome_iso_kernel(const _Float16* __restrict__ xn,
                                                         const float* __restrict__ margin_p,
                                                         float* __restrict__ iso_part) {
    __shared__ __align__(16) _Float16 Bs[2][16 * C_];   // 2 x 24KB

    const int tilesPerBatch = T_ / 128;
    const int s    = blockIdx.x % NSPLIT_;
    const int rest = blockIdx.x / NSPLIT_;
    const int b    = rest / tilesPerBatch;
    const int tile = rest % tilesPerBatch;

    const int tid  = threadIdx.x;
    const int wave = tid >> 5;
    const int lane = tid & 31;
    const int m    = lane & 15;     // row / column within tile for this lane
    const int hs   = lane >> 4;     // half-select
    const int row_base = tile * 128 + wave * 16;
    const int n0   = s * NCHUNK_;

    const _Float16* base = xn + (size_t)b * T_ * C_;
    const _Float16* arow = base + (size_t)(row_base + m) * C_;
    const float margin = *margin_p;

    // Cooperative stage of one 16-column B tile (contiguous 24KB) into LDS.
    auto stage = [&](int buf, int nt) {
        const _Float16* src = base + (size_t)nt * C_;   // 16 rows contiguous
#pragma unroll
        for (int e = 0; e < 6; ++e) {
            const int idx = (tid + e * 256) * 8;        // half index, 16B chunks
            *(v8h*)(Bs[buf] + idx) = *(const v8h*)(src + idx);
        }
    };

    float rowsum[8];
#pragma unroll
    for (int r = 0; r < 8; ++r) rowsum[r] = 0.f;

    stage(0, n0);
    __syncthreads();

    for (int nt = n0; nt < n0 + NCHUNK_; nt += 16) {
        const int cur = (nt >> 4) & 1;
        if (nt + 16 < n0 + NCHUNK_) stage(cur ^ 1, nt + 16);

        const _Float16* bl = Bs[cur] + (size_t)m * C_;  // lane's column row in LDS
        v8f acc = {};
#pragma unroll
        for (int k = 0; k < C_; k += 32) {
            // A fragment (16x32 f16): lane holds row m, 16 contiguous K.
            // Loop-invariant over nt -> compiler keeps all 24 fragments in VGPRs.
            v16h a = *(const v16h*)(arow + k + hs * 16);
            // B fragment (32x16 f16): lane holds column m, 16 contiguous K, from LDS
            v16h bb = *(const v16h*)(bl + k + hs * 16);
            acc = __builtin_amdgcn_wmma_f32_16x16x32_f16(
                false, a, false, bb, (short)0, acc, false, false);
        }
        // elementwise elu((d - margin)/0.01) and row accumulation
#pragma unroll
        for (int r = 0; r < 8; ++r) {
            float z = (acc[r] - margin) * 100.0f;
            rowsum[r] += (z > 0.f) ? z : (__expf(z) - 1.0f);
        }
        __syncthreads();    // reads of cur done; stage writes to cur^1 visible
    }

    // Reduce across the 16 lanes of each half (xor masks < 16 keep bit4 fixed)
#pragma unroll
    for (int r = 0; r < 8; ++r) {
        float v = rowsum[r];
        for (int off = 1; off < 16; off <<= 1) v += __shfl_xor(v, off, 32);
        rowsum[r] = v;
    }
    if ((lane & 15) == 0) {
        const int rb = row_base + hs * 8;   // C/D layout: VGPR r -> row r + 8*hs
        float* dst = iso_part + (size_t)s * (B_ * T_) + (size_t)b * T_ + rb;
#pragma unroll
        for (int r = 0; r < 8; ++r) dst[r] = rowsum[r];
    }
}

// 2b) Combine partial sums -> iso (mean over T)
__global__ __launch_bounds__(256) void pitome_iso_combine_kernel(const float* __restrict__ part,
                                                                 float* __restrict__ iso) {
    const int i = blockIdx.x * 256 + threadIdx.x;   // over B*T
    float s = 0.f;
#pragma unroll
    for (int p = 0; p < NSPLIT_; ++p) s += part[(size_t)p * (B_ * T_) + i];
    iso[i] = s * (1.0f / (float)T_);
}

// ---------------------------------------------------------------------------
// 3) Per-batch bitonic argsort: descending by iso, ties ascending by index
// ---------------------------------------------------------------------------
__global__ __launch_bounds__(1024) void pitome_sort_kernel(const float* __restrict__ iso,
                                                           int* __restrict__ order) {
    __shared__ float sv[T_];
    __shared__ int   si[T_];
    const int b = blockIdx.x;
    for (int i = threadIdx.x; i < T_; i += 1024) {
        sv[i] = iso[(size_t)b * T_ + i];
        si[i] = i;
    }
    __syncthreads();
    for (int k = 2; k <= T_; k <<= 1) {
        for (int j = k >> 1; j > 0; j >>= 1) {
            for (int i = threadIdx.x; i < T_; i += 1024) {
                const int ixj = i ^ j;
                if (ixj > i) {
                    float v1 = sv[i], v2 = sv[ixj];
                    int   i1 = si[i], i2 = si[ixj];
                    bool beforeIxj = (v2 > v1) || (v2 == v1 && i2 < i1);
                    bool fwd = ((i & k) == 0);
                    if (fwd ? beforeIxj : !beforeIxj) {
                        sv[i] = v2; sv[ixj] = v1;
                        si[i] = i2; si[ixj] = i1;
                    }
                }
            }
            __syncthreads();
        }
    }
    for (int i = threadIdx.x; i < T_; i += 1024) order[(size_t)b * T_ + i] = si[i];
}

// ---------------------------------------------------------------------------
// 4) w = 1 - softmax(iso) per batch
// ---------------------------------------------------------------------------
__global__ __launch_bounds__(1024) void pitome_softmax_kernel(const float* __restrict__ iso,
                                                              float* __restrict__ w) {
    __shared__ float red[1024];
    const int b = blockIdx.x, tid = threadIdx.x;
    const float* ib = iso + (size_t)b * T_;
    float mx = -1e30f;
    for (int i = tid; i < T_; i += 1024) mx = fmaxf(mx, ib[i]);
    red[tid] = mx; __syncthreads();
    for (int off = 512; off > 0; off >>= 1) {
        if (tid < off) red[tid] = fmaxf(red[tid], red[tid + off]);
        __syncthreads();
    }
    mx = red[0]; __syncthreads();
    float sm = 0.f;
    for (int i = tid; i < T_; i += 1024) sm += __expf(ib[i] - mx);
    red[tid] = sm; __syncthreads();
    for (int off = 512; off > 0; off >>= 1) {
        if (tid < off) red[tid] += red[tid + off];
        __syncthreads();
    }
    sm = red[0]; __syncthreads();
    float* wb = w + (size_t)b * T_;
    for (int i = tid; i < T_; i += 1024) wb[i] = 1.0f - __expf(ib[i] - mx) / sm;
}

// ---------------------------------------------------------------------------
// 5) dst_idx[b][a] = argmax_b' dot(xn[a_idx[a]], xn[b_idx[b']])
//    (elu((.-m)/0.01) is monotone -> argmax over raw dots). One block per (b,a).
// ---------------------------------------------------------------------------
__global__ __launch_bounds__(256) void pitome_dst_kernel(const _Float16* __restrict__ xn,
                                                         const int* __restrict__ order,
                                                         int* __restrict__ dst_idx) {
    __shared__ _Float16 arow_s[C_];
    __shared__ float rv[256];
    __shared__ int   ri[256];
    const int b = blockIdx.x / R_;
    const int a = blockIdx.x % R_;
    const int tid = threadIdx.x;
    const int* ob = order + (size_t)b * T_;
    const _Float16* xb = xn + (size_t)b * T_ * C_;

    const _Float16* ar = xb + (size_t)ob[a] * C_;
#pragma unroll
    for (int e = 0; e < 3; ++e) arow_s[tid + e * 256] = ar[tid + e * 256];
    __syncthreads();

    float best = -1e30f; int besti = 0x7fffffff;
    if (tid < R_) {
        const _Float16* br = xb + (size_t)ob[R_ + tid] * C_;
        float acc = 0.f;
#pragma unroll 8
        for (int c = 0; c < C_; ++c) acc += (float)arow_s[c] * (float)br[c];
        best = acc; besti = tid;
    }
    rv[tid] = best; ri[tid] = besti;
    __syncthreads();
    for (int off = 128; off > 0; off >>= 1) {
        if (tid < off) {
            if (rv[tid + off] > rv[tid] ||
                (rv[tid + off] == rv[tid] && ri[tid + off] < ri[tid])) {
                rv[tid] = rv[tid + off]; ri[tid] = ri[tid + off];
            }
        }
        __syncthreads();
    }
    if (tid == 0) dst_idx[(size_t)b * R_ + a] = ri[0];
}

// ---------------------------------------------------------------------------
// 6) Deterministic merge accumulation per destination j (no float atomics)
// ---------------------------------------------------------------------------
__global__ __launch_bounds__(256) void pitome_accum_kernel(const float* __restrict__ x,
                                                           const float* __restrict__ mask,
                                                           const float* __restrict__ w,
                                                           const int* __restrict__ order,
                                                           const int* __restrict__ dst_idx,
                                                           float* __restrict__ accx,
                                                           float* __restrict__ accs,
                                                           float* __restrict__ accm) {
    const int b = blockIdx.x / R_;
    const int j = blockIdx.x % R_;
    const int tid = threadIdx.x;
    const int* ob = order + (size_t)b * T_;
    const float* xb = x + (size_t)b * T_ * C_;
    const float* wb = w + (size_t)b * T_;
    const float* mb = mask + (size_t)b * T_;

    const int bj = ob[R_ + j];
    const float wd = wb[bj];
    float size = wd;
    float mm = mb[bj];
    float acc[3];
#pragma unroll
    for (int e = 0; e < 3; ++e) acc[e] = xb[(size_t)bj * C_ + tid + e * 256] * wd;

    for (int a = 0; a < R_; ++a) {
        if (dst_idx[(size_t)b * R_ + a] == j) {
            const int ar = ob[a];
            const float ws = wb[ar];
#pragma unroll
            for (int e = 0; e < 3; ++e)
                acc[e] += xb[(size_t)ar * C_ + tid + e * 256] * ws;
            size += ws;
            mm = fmaxf(mm, mb[ar]);
        }
    }
    float* ax = accx + ((size_t)b * R_ + j) * C_;
#pragma unroll
    for (int e = 0; e < 3; ++e) ax[tid + e * 256] = acc[e];
    if (tid == 0) {
        accs[(size_t)b * R_ + j] = size;
        accm[(size_t)b * R_ + j] = mm;
    }
}

// ---------------------------------------------------------------------------
// 7) Final outputs: xm (B,TOUT,C) then am (B,TOUT), flat in d_out.
//    Protected rows: (x*w)/w == x exactly.
// ---------------------------------------------------------------------------
__global__ __launch_bounds__(256) void pitome_out_kernel(const float* __restrict__ x,
                                                         const float* __restrict__ mask,
                                                         const int* __restrict__ order,
                                                         const float* __restrict__ accx,
                                                         const float* __restrict__ accs,
                                                         const float* __restrict__ accm,
                                                         float* __restrict__ out) {
    const int b = blockIdx.x / TOUT_;
    const int p = blockIdx.x % TOUT_;
    const int tid = threadIdx.x;
    float* xm = out + ((size_t)b * TOUT_ + p) * C_;
    float* am = out + (size_t)B_ * TOUT_ * C_ + (size_t)b * TOUT_ + p;

    if (p < NPROT_) {
        const int src = order[(size_t)b * T_ + 2 * R_ + p];
        const float* xr = x + ((size_t)b * T_ + src) * C_;
#pragma unroll
        for (int e = 0; e < 3; ++e) xm[tid + e * 256] = xr[tid + e * 256];
        if (tid == 0) *am = mask[(size_t)b * T_ + src];
    } else {
        const int j = p - NPROT_;
        const float inv = 1.0f / accs[(size_t)b * R_ + j];
        const float* ax = accx + ((size_t)b * R_ + j) * C_;
#pragma unroll
        for (int e = 0; e < 3; ++e) xm[tid + e * 256] = ax[tid + e * 256] * inv;
        if (tid == 0) *am = accm[(size_t)b * R_ + j];
    }
}

// ---------------------------------------------------------------------------
extern "C" void kernel_launch(void* const* d_in, const int* in_sizes, int n_in,
                              void* d_out, int out_size, void* d_ws, size_t ws_size,
                              hipStream_t stream) {
    const float* x      = (const float*)d_in[0];
    const float* mask   = (const float*)d_in[1];
    const float* margin = (const float*)d_in[2];

    char* ws = (char*)d_ws;
    size_t off = 0;
    auto carve = [&](size_t bytes) -> void* {
        void* p = ws + off;
        off += (bytes + 255) & ~(size_t)255;
        return p;
    };
    _Float16* xn   = (_Float16*)carve((size_t)B_ * T_ * C_ * sizeof(_Float16));
    float*    isop = (float*)   carve((size_t)NSPLIT_ * B_ * T_ * sizeof(float));
    float*    iso  = (float*)   carve((size_t)B_ * T_ * sizeof(float));
    int*      ordr = (int*)     carve((size_t)B_ * T_ * sizeof(int));
    float*    w    = (float*)   carve((size_t)B_ * T_ * sizeof(float));
    int*      dsti = (int*)     carve((size_t)B_ * R_ * sizeof(int));
    float*    accx = (float*)   carve((size_t)B_ * R_ * C_ * sizeof(float));
    float*    accs = (float*)   carve((size_t)B_ * R_ * sizeof(float));
    float*    accm = (float*)   carve((size_t)B_ * R_ * sizeof(float));

    pitome_norm_kernel<<<B_ * T_, 256, 0, stream>>>(x, xn);
    pitome_iso_kernel<<<B_ * (T_ / 128) * NSPLIT_, 256, 0, stream>>>(xn, margin, isop);
    pitome_iso_combine_kernel<<<(B_ * T_) / 256, 256, 0, stream>>>(isop, iso);
    pitome_sort_kernel<<<B_, 1024, 0, stream>>>(iso, ordr);
    pitome_softmax_kernel<<<B_, 1024, 0, stream>>>(iso, w);
    pitome_dst_kernel<<<B_ * R_, 256, 0, stream>>>(xn, ordr, dsti);
    pitome_accum_kernel<<<B_ * R_, 256, 0, stream>>>(x, mask, w, ordr, dsti,
                                                     accx, accs, accm);
    pitome_out_kernel<<<B_ * TOUT_, 256, 0, stream>>>(x, mask, ordr,
                                                      accx, accs, accm,
                                                      (float*)d_out);
}
